// Head_4733053960485
// MI455X (gfx1250) — compile-verified, matrix operand order checked by reading, not run on hardware
//
#include <hip/hip_runtime.h>

// Fused causal attention head for MI455X (gfx1250, wave32, WMMA).
// One workgroup (256 threads = 8 wave32) per batch element.
// Precision: bf16 WMMA inputs, f32 accumulation (V_WMMA_F32_16X16X32_BF16).

#define BATCH_N 1024
#define SEQ_T   256
#define EMB_C   384
#define HEAD_H  64

typedef __attribute__((ext_vector_type(16))) __bf16 v16bf;
typedef __attribute__((ext_vector_type(8)))  __bf16 v8bf;
typedef __attribute__((ext_vector_type(8)))  float  v8f;
typedef __attribute__((ext_vector_type(4)))  float  v4f;

static __device__ __forceinline__ v16bf cat8(v8bf lo, v8bf hi) {
  return __builtin_shufflevector(lo, hi, 0,1,2,3,4,5,6,7,8,9,10,11,12,13,14,15);
}
// Load one 16-bit A/B fragment: two runs of 8 contiguous bf16, 16 elements apart
// (matches the CDNA5 16-bit A/B VGPR layout: K = 8*half+{0..7} and 16+8*half+{0..7}).
static __device__ __forceinline__ v16bf lds_frag(const __bf16* p) {
  return cat8(*(const v8bf*)p, *(const v8bf*)(p + 16));
}
static __device__ __forceinline__ v8f wmma_bf16(v16bf a, v16bf b, v8f c) {
  // (neg_a, A, neg_b, B, c_mod, C, reuse_a, reuse_b)
  return __builtin_amdgcn_wmma_f32_16x16x32_bf16(false, a, false, b, (short)0, c,
                                                 false, false);
}

__global__ __launch_bounds__(256)
void head_attn_kernel(const float* __restrict__ x,
                      const float* __restrict__ Wq,
                      const float* __restrict__ Wk,
                      const float* __restrict__ Wv,
                      float* __restrict__ out)
{
  // 48K + 32K + 32K + 32K + 8K = 152 KB LDS (fits CU-mode 160 KB cap)
  __shared__ __align__(16) __bf16 sWT[HEAD_H * EMB_C];   // W^T  [64][384] bf16
  __shared__ __align__(16) __bf16 sQ [SEQ_T * HEAD_H];   // Q    [256][64] bf16
  __shared__ __align__(16) __bf16 sK [SEQ_T * HEAD_H];   // K    [256][64] bf16
  __shared__ __align__(16) __bf16 sVT[HEAD_H * SEQ_T];   // V^T  [64][256] bf16
  __shared__ __align__(16) __bf16 sP [8 * 16 * 32];      // per-wave P scratch

  const int b    = blockIdx.x;
  const int tid  = threadIdx.x;
  const int wv   = tid >> 5;
  const int lane = tid & 31;
  const int hf   = lane >> 4;   // half-wave select
  const int l16  = lane & 15;

  const float* xb = x + (size_t)b * SEQ_T * EMB_C;
  const v8f vzero = {0.f,0.f,0.f,0.f,0.f,0.f,0.f,0.f};

  // ---------------- Phase 1: Q/K/V projections (x @ W) ----------------
  for (int p = 0; p < 3; ++p) {
    const float* W = (p == 0) ? Wq : ((p == 1) ? Wk : Wv);
    __syncthreads();                       // previous consumers of sWT done
    for (int i = tid; i < EMB_C * HEAD_H; i += 256) {
      const int c = i >> 6, n = i & 63;    // W is [384][64] row-major
      sWT[n * EMB_C + c] = (__bf16)W[i];   // store transposed, bf16
    }
    __syncthreads();

    for (int rep = 0; rep < 2; ++rep) {
      const int rt = wv + rep * 8;         // this wave's 16-row tile
      const int t0 = rt * 16;
      v8f acc[4];
      #pragma unroll
      for (int nt = 0; nt < 4; ++nt) acc[nt] = vzero;

      const float* xr = xb + (size_t)(t0 + l16) * EMB_C + hf * 8;
      #pragma unroll
      for (int s = 0; s < 12; ++s) {       // K-dim 384 in steps of 32
        v4f a0 = *(const v4f*)(xr + 0);    // features cbase + 8*hf + {0..7}
        v4f a1 = *(const v4f*)(xr + 4);
        v4f a2 = *(const v4f*)(xr + 16);   // features cbase + 16 + 8*hf + {0..7}
        v4f a3 = *(const v4f*)(xr + 20);
        v16bf afrag;
        #pragma unroll
        for (int e = 0; e < 4; ++e) {
          afrag[e]      = (__bf16)a0[e];
          afrag[4 + e]  = (__bf16)a1[e];
          afrag[8 + e]  = (__bf16)a2[e];
          afrag[12 + e] = (__bf16)a3[e];
        }
        const int cbase = s * 32;
        #pragma unroll
        for (int nt = 0; nt < 4; ++nt) {
          v16bf bfrag = lds_frag(&sWT[(nt * 16 + l16) * EMB_C + cbase + hf * 8]);
          acc[nt] = wmma_bf16(afrag, bfrag, acc[nt]);
        }
        xr += 32;
        if (s + 1 < 12) __builtin_prefetch(xr, 0, 3);  // global_prefetch_b8
      }

      if (p == 2) {
        // V stored transposed: C/D layout rows are consecutive per lane ->
        // one packed ds_store_b128 per tile.
        #pragma unroll
        for (int nt = 0; nt < 4; ++nt) {
          v8bf pk;
          #pragma unroll
          for (int g = 0; g < 8; ++g) pk[g] = (__bf16)acc[nt][g];
          *(v8bf*)&sVT[(nt * 16 + l16) * SEQ_T + t0 + hf * 8] = pk;
        }
      } else {
        __bf16* dst = (p == 0) ? sQ : sK;  // row-major [256][64]
        #pragma unroll
        for (int nt = 0; nt < 4; ++nt) {
          const int col = nt * 16 + l16;
          #pragma unroll
          for (int g = 0; g < 8; ++g)
            dst[(t0 + hf * 8 + g) * HEAD_H + col] = (__bf16)acc[nt][g];
        }
      }
    }
  }
  __syncthreads();

  // ---------------- Phase 2: flash attention over K/V ----------------
  const float scale = 0.051031036307982884f;   // 1/sqrt(384) (note: C, not H)
  float* outb = out + (size_t)b * SEQ_T * HEAD_H;
  __bf16* myP = &sP[wv * 512];

  for (int rep = 0; rep < 2; ++rep) {
    const int r  = rep ? (15 - wv) : wv;       // triangular load balance
    const int t0 = r * 16;

    const __bf16* qr = &sQ[(t0 + l16) * HEAD_H + hf * 8];
    v16bf qa0 = lds_frag(qr);                  // features 0..31
    v16bf qa1 = lds_frag(qr + 32);             // features 32..63

    v8f o[4];
    #pragma unroll
    for (int nt = 0; nt < 4; ++nt) o[nt] = vzero;
    float mrun[8], lrun[8];
    #pragma unroll
    for (int g = 0; g < 8; ++g) { mrun[g] = -3.0e38f; lrun[g] = 0.f; }

    const int nblk = (r >> 1) + 1;             // causal: 32-key blocks
    for (int jb = 0; jb < nblk; ++jb) {
      const int k0 = jb * 32;
      v8f s0 = vzero, s1 = vzero;
      {
        const __bf16* kr0 = &sK[(k0      + l16) * HEAD_H + hf * 8];
        const __bf16* kr1 = &sK[(k0 + 16 + l16) * HEAD_H + hf * 8];
        s0 = wmma_bf16(qa0, lds_frag(kr0),      s0);
        s0 = wmma_bf16(qa1, lds_frag(kr0 + 32), s0);
        s1 = wmma_bf16(qa0, lds_frag(kr1),      s1);
        s1 = wmma_bf16(qa1, lds_frag(kr1 + 32), s1);
      }

      const int kc0 = k0 + l16;
      const int kc1 = kc0 + 16;
      #pragma unroll
      for (int g = 0; g < 8; ++g) {
        const int q = t0 + hf * 8 + g;
        float a = (kc0 <= q) ? s0[g] * scale : -3.0e38f;  // causal mask
        float c = (kc1 <= q) ? s1[g] * scale : -3.0e38f;
        // row-max across the 16-lane N group (C/D layout: N = lane%16)
        float mx = fmaxf(a, c);
        mx = fmaxf(mx, __shfl_xor(mx, 1, 32));
        mx = fmaxf(mx, __shfl_xor(mx, 2, 32));
        mx = fmaxf(mx, __shfl_xor(mx, 4, 32));
        mx = fmaxf(mx, __shfl_xor(mx, 8, 32));
        const float mnew = fmaxf(mrun[g], mx);
        const float f  = __expf(mrun[g] - mnew);
        const float p0 = __expf(a - mnew);
        const float p1 = __expf(c - mnew);
        float rs = p0 + p1;
        rs += __shfl_xor(rs, 1, 32);
        rs += __shfl_xor(rs, 2, 32);
        rs += __shfl_xor(rs, 4, 32);
        rs += __shfl_xor(rs, 8, 32);
        lrun[g] = lrun[g] * f + rs;
        mrun[g] = mnew;
        #pragma unroll
        for (int nt = 0; nt < 4; ++nt) o[nt][g] *= f;   // rescale O
        s0[g] = p0; s1[g] = p1;
      }

      // C/D-layout -> A-fragment layout for P via per-wave LDS scratch
      {
        const int mb = hf * 8;
        #pragma unroll
        for (int g = 0; g < 8; ++g) {
          myP[(mb + g) * 32 +      l16] = (__bf16)s0[g];
          myP[(mb + g) * 32 + 16 + l16] = (__bf16)s1[g];
        }
      }
      asm volatile("s_wait_dscnt 0" ::: "memory");      // wave-local RAW on LDS
      v16bf pa = lds_frag(&myP[l16 * 32 + hf * 8]);

      #pragma unroll
      for (int nt = 0; nt < 4; ++nt) {
        v16bf vb = lds_frag(&sVT[(nt * 16 + l16) * SEQ_T + k0 + hf * 8]);
        o[nt] = wmma_bf16(pa, vb, o[nt]);
      }
    }

    // epilogue: divide by softmax denominator, store f32
    #pragma unroll
    for (int nt = 0; nt < 4; ++nt) {
      #pragma unroll
      for (int g = 0; g < 8; ++g) {
        outb[(t0 + hf * 8 + g) * HEAD_H + nt * 16 + l16] = o[nt][g] / lrun[g];
      }
    }
  }
}

extern "C" void kernel_launch(void* const* d_in, const int* in_sizes, int n_in,
                              void* d_out, int out_size, void* d_ws, size_t ws_size,
                              hipStream_t stream) {
  (void)in_sizes; (void)n_in; (void)out_size; (void)d_ws; (void)ws_size;
  const float* x  = (const float*)d_in[0];
  const float* Wq = (const float*)d_in[1];
  const float* Wk = (const float*)d_in[2];
  const float* Wv = (const float*)d_in[3];
  float* out = (float*)d_out;
  head_attn_kernel<<<BATCH_N, 256, 0, stream>>>(x, Wq, Wk, Wv, out);
}
